// LSTM_27874337751637
// MI455X (gfx1250) — matrix operand order repaired and checked by reading, not compiled, not run on hardware
//
#include <hip/hip_runtime.h>

// ---------------------------------------------------------------------------
// GraphLSTM-GCN fused pipeline for gfx1250 (MI455X), wave32 + WMMA bf16.
//   gates = x @ w_ih.T + b_ih + b_hh ; h = sig(o)*tanh(sig(i)*tanh(g))
//   m = h @ w_gcn.T ; out = relu(scatter_add(m[src]*norm, dst) + b_gcn)
// Weights are pre-converted to bf16 in workspace once (L2-resident), so the
// WMMA loops issue pure b128 loads with no per-use f32->bf16 conversion.
// ---------------------------------------------------------------------------

typedef __attribute__((ext_vector_type(16))) __bf16 v16bf;
typedef __attribute__((ext_vector_type(8)))  float  v8f;

#define DIM      128            // IN == H == OUT == 128
#define GDIM     512            // 4*H
#define ROWS_PER_WAVE 16
#define WAVES_PER_BLK 8
#define ROWS_PER_BLK  (ROWS_PER_WAVE * WAVES_PER_BLK)   // 128 nodes / block

// A-matrix 16x32 bf16 lane layout: lane m=lane&15,
// element e -> K = kk + e + 16*(lane>=16)
__device__ __forceinline__ v16bf load_a_f32(const float* __restrict__ row,
                                            int kk, int half) {
  v16bf a;
  const int base = kk + 16 * half;
#pragma unroll
  for (int e = 0; e < 16; ++e) {
    const int k = base + e;
    a[e] = (__bf16)row[k];
  }
  return a;
}

__device__ __forceinline__ v16bf load_a_lds(const __bf16* __restrict__ row,
                                            int kk, int half) {
  v16bf a;
  const int base = kk + 16 * half;
#pragma unroll
  for (int e = 0; e < 16; ++e) {
    const int k = base + e;
    a[e] = row[k];
  }
  return a;
}

// B-matrix 32x16 bf16: lane n=lane&15, element e -> K = kk + e + 16*(lane>=16)
// B[k][n] = W[colbase+n][k]  (W row-major [cols][128] bf16, 32B-aligned rows)
__device__ __forceinline__ v16bf load_b_bf(const __bf16* __restrict__ W,
                                           int colbase, int n, int kk, int half) {
  const __bf16* wrow = W + (size_t)(colbase + n) * DIM + kk + 16 * half;
  return *(const v16bf*)wrow;    // 32 bytes, 32B-aligned -> 2x global_load_b128
}

__device__ __forceinline__ v8f wmma_bf16(v16bf a, v16bf b, v8f c) {
  return __builtin_amdgcn_wmma_f32_16x16x32_bf16(
      /*neg_a=*/false, a, /*neg_b=*/false, b,
      /*c_mod=*/(short)0, c, /*reuse_a=*/false, /*reuse_b=*/false);
}

__device__ __forceinline__ float sigf(float v) {
  return 1.0f / (1.0f + __expf(-v));
}

// --------------------------- kernels ---------------------------------------

__global__ void k_init(float* __restrict__ deg, float* __restrict__ out,
                       int n, int total) {
  int i = blockIdx.x * blockDim.x + threadIdx.x;
  if (i < total) out[i] = 0.0f;
  if (i < n)     deg[i] = 1.0f;   // self-loop contribution
}

// One-shot f32 -> bf16 weight conversion into workspace.
__global__ void k_prep(const float* __restrict__ w_ih,
                       const float* __restrict__ w_gcn,
                       __bf16* __restrict__ wih_bf,
                       __bf16* __restrict__ wgcn_bf) {
  int i = blockIdx.x * blockDim.x + threadIdx.x;
  const int n_ih = GDIM * DIM;                 // 65536
  if (i < n_ih) wih_bf[i] = (__bf16)w_ih[i];
  else {
    int j = i - n_ih;                          // 0 .. 16383
    wgcn_bf[j] = (__bf16)w_gcn[j];
  }
}

__global__ void k_deg(const int* __restrict__ ei, float* __restrict__ deg,
                      int E) {
  int e = blockIdx.x * blockDim.x + threadIdx.x;
  if (e < E) atomicAdd(&deg[ei[E + e]], 1.0f);   // dst row of edge_index
}

__global__ void k_rsqrt(float* __restrict__ deg, int n) {
  int i = blockIdx.x * blockDim.x + threadIdx.x;
  if (i < n) deg[i] = rsqrtf(deg[i]);            // deg >= 1 always
}

// Fused LSTM + GCN-linear node transform.  One wave owns a 16-node tile.
__global__ __launch_bounds__(256) void
k_transform(const float* __restrict__ x,
            const __bf16* __restrict__ wih_bf,
            const float* __restrict__ b_ih,
            const float* __restrict__ b_hh,
            const __bf16* __restrict__ wgcn_bf,
            float* __restrict__ m_out, int N) {
  __shared__ __bf16 lds_h[WAVES_PER_BLK][ROWS_PER_WAVE][DIM];   // 32 KB

  const int tid  = threadIdx.x;
  const int wave = tid >> 5;
  const int lane = tid & 31;
  const int half = lane >> 4;       // which 16-lane group
  const int nn   = lane & 15;       // A-row / B-col / D-col index
  const int rowBase = blockIdx.x * ROWS_PER_BLK + wave * ROWS_PER_WAVE;

  // ---- stage A (x rows, 16x128) into registers as 4 bf16 k-chunks ----
  int arow = rowBase + nn;
  if (arow >= N) arow = N - 1;      // clamp: WMMA needs full EXEC
  const float* xrow = x + (size_t)arow * DIM;
  v16bf axc[4];
#pragma unroll
  for (int c = 0; c < 4; ++c) axc[c] = load_a_f32(xrow, c * 32, half);

  // ---- GEMM1 + activations, per 16-col tile of H ----
  for (int ct = 0; ct < 8; ++ct) {
    const int colI = 0 * DIM + ct * 16;     // i gate
    const int colG = 2 * DIM + ct * 16;     // g gate
    const int colO = 3 * DIM + ct * 16;     // o gate   (f gate unused: c0=0)

    const float biasI = b_ih[colI + nn] + b_hh[colI + nn];
    const float biasG = b_ih[colG + nn] + b_hh[colG + nn];
    const float biasO = b_ih[colO + nn] + b_hh[colO + nn];
    v8f accI, accG, accO;
#pragma unroll
    for (int r = 0; r < 8; ++r) { accI[r] = biasI; accG[r] = biasG; accO[r] = biasO; }

#pragma unroll
    for (int c = 0; c < 4; ++c) {
      const int kk = c * 32;
      accI = wmma_bf16(axc[c], load_b_bf(wih_bf, colI, nn, kk, half), accI);
      accG = wmma_bf16(axc[c], load_b_bf(wih_bf, colG, nn, kk, half), accG);
      accO = wmma_bf16(axc[c], load_b_bf(wih_bf, colO, nn, kk, half), accO);
    }

    // h = sig(o) * tanh( sig(i) * tanh(g) ), purely on accumulator lanes
#pragma unroll
    for (int r = 0; r < 8; ++r) {
      const float cval = sigf(accI[r]) * tanhf(accG[r]);
      const float h    = sigf(accO[r]) * tanhf(cval);
      lds_h[wave][r + 8 * half][ct * 16 + nn] = (__bf16)h;
    }
  }

  __syncthreads();   // wave-private tiles; barrier is belt-and-braces

  // ---- GEMM2: m = h @ w_gcn.T ----
  const __bf16* hrow = &lds_h[wave][nn][0];
  for (int ot = 0; ot < 8; ++ot) {
    v8f acc = {};
#pragma unroll
    for (int c = 0; c < 4; ++c) {
      const int kk = c * 32;
      acc = wmma_bf16(load_a_lds(hrow, kk, half),
                      load_b_bf(wgcn_bf, ot * 16, nn, kk, half), acc);
    }
#pragma unroll
    for (int r = 0; r < 8; ++r) {
      const int node = rowBase + r + 8 * half;
      if (node < N) m_out[(size_t)node * DIM + ot * 16 + nn] = acc[r];
    }
  }
}

// Edge scatter: 32 lanes per edge, float4 per lane, L2-resident atomics.
__global__ __launch_bounds__(256) void
k_scatter(const int* __restrict__ ei, const float* __restrict__ m,
          const float* __restrict__ dis, float* __restrict__ out, int E) {
  const int t    = blockIdx.x * blockDim.x + threadIdx.x;
  const int e    = t >> 5;
  const int lane = t & 31;
  if (e >= E) return;

  // lookahead prefetch of a future edge's message row (global_prefetch_b8)
  const int ep = e + 256;
  if (ep < E) {
    const int psrc = ei[ep];
    __builtin_prefetch(m + (size_t)psrc * DIM + lane * 4, 0, 0);
  }

  const int src = ei[e];
  const int dst = ei[E + e];
  const float w = dis[src] * dis[dst];

  const float4 v = ((const float4*)(m + (size_t)src * DIM))[lane];
  float* op = out + (size_t)dst * DIM + lane * 4;
  atomicAdd(op + 0, v.x * w);
  atomicAdd(op + 1, v.y * w);
  atomicAdd(op + 2, v.z * w);
  atomicAdd(op + 3, v.w * w);
}

// Self-loop term + bias + ReLU.
__global__ void k_finalize(float* __restrict__ out, const float* __restrict__ m,
                           const float* __restrict__ dis,
                           const float* __restrict__ b_gcn, int total) {
  int i = blockIdx.x * blockDim.x + threadIdx.x;
  if (i >= total) return;
  const int n = i >> 7;          // / DIM
  const int c = i & (DIM - 1);
  const float d = dis[n];
  const float v = out[i] + m[i] * d * d + b_gcn[c];
  out[i] = fmaxf(v, 0.0f);
}

// --------------------------- launcher --------------------------------------

extern "C" void kernel_launch(void* const* d_in, const int* in_sizes, int n_in,
                              void* d_out, int out_size, void* d_ws, size_t ws_size,
                              hipStream_t stream) {
  const float* x     = (const float*)d_in[0];
  const int*   ei    = (const int*)  d_in[1];
  const float* w_ih  = (const float*)d_in[2];
  // d_in[3] = w_hh : unused (h0 = c0 = 0 for a single timestep)
  const float* b_ih  = (const float*)d_in[4];
  const float* b_hh  = (const float*)d_in[5];
  const float* w_gcn = (const float*)d_in[6];
  const float* b_gcn = (const float*)d_in[7];
  float* out = (float*)d_out;

  const int N = in_sizes[0] / DIM;
  const int E = in_sizes[1] / 2;
  const int total = N * DIM;

  // workspace layout (floats): [deg: N][pad to 64K][m: N*128][bf16 weights]
  float* ws  = (float*)d_ws;
  float* deg = ws;                                   // N floats -> dis
  float* m   = ws + (1 << 16);                       // 256 KB offset
  __bf16* wih_bf  = (__bf16*)(m + (size_t)N * DIM);  // 512*128 bf16 (128 KB)
  __bf16* wgcn_bf = wih_bf + GDIM * DIM;             // 128*128 bf16 (32 KB)

  k_init<<<(total + 255) / 256, 256, 0, stream>>>(deg, out, N, total);
  k_prep<<<(GDIM * DIM + DIM * DIM + 255) / 256, 256, 0, stream>>>(
      w_ih, w_gcn, wih_bf, wgcn_bf);
  k_deg<<<(E + 255) / 256, 256, 0, stream>>>(ei, deg, E);
  k_rsqrt<<<(N + 255) / 256, 256, 0, stream>>>(deg, N);

  const int nblk = (N + ROWS_PER_BLK - 1) / ROWS_PER_BLK;
  k_transform<<<nblk, 256, 0, stream>>>(x, wih_bf, b_ih, b_hh, wgcn_bf, m, N);

  const long long sthreads = (long long)E * 32;
  k_scatter<<<(int)((sthreads + 255) / 256), 256, 0, stream>>>(ei, m, deg, out, E);

  k_finalize<<<(total + 255) / 256, 256, 0, stream>>>(out, m, deg, b_gcn, total);
}
